// SwinTransformer_18202071401050
// MI455X (gfx1250) — compile-verified, matrix operand order checked by reading, not run on hardware
//
#include <hip/hip_runtime.h>
#include <hip/hip_bf16.h>

// ---------------------------------------------------------------------------
// CDNA5 (gfx1250) Swin-Transformer forward.
//  - All dense matmuls + attention matmuls via v_wmma_f32_16x16x32_bf16.
//  - Weights pre-converted once per launch to padded bf16; GEMM B-tiles are
//    fetched with gfx1250 async-to-LDS copies (global_load_async_to_lds_b128,
//    s_wait_asynccnt) so the fill overlaps the A-tile f32->bf16 conversion.
// ---------------------------------------------------------------------------

typedef __attribute__((ext_vector_type(16))) __bf16 v16bf;
typedef __attribute__((ext_vector_type(8)))  float  v8f;

__device__ inline unsigned short f2bf(float x) {
    unsigned int u = __float_as_uint(x);
    return (unsigned short)((u + 0x7FFFu + ((u >> 16) & 1u)) >> 16);  // RNE
}
__device__ inline unsigned pack_bf2(float a, float b) {
    return (unsigned)f2bf(a) | ((unsigned)f2bf(b) << 16);
}

union FragU { unsigned int u[8]; v16bf v; };

// Load one 16x32 bf16 WMMA operand fragment for this lane. `p` points at this
// lane's 32-element K-window (K contiguous); h = lane/16 selects the K-phase:
//   elems 0..7  -> K = h*8 + 0..7, elems 8..15 -> K = 16 + h*8 + 0..7
__device__ inline v16bf load_frag(const unsigned short* p, int h) {
    const uint4* rp = (const uint4*)p;
    uint4 lo = rp[h];
    uint4 hi = rp[2 + h];
    FragU f;
    f.u[0] = lo.x; f.u[1] = lo.y; f.u[2] = lo.z; f.u[3] = lo.w;
    f.u[4] = hi.x; f.u[5] = hi.y; f.u[6] = hi.z; f.u[7] = hi.w;
    return f.v;
}

__device__ inline v8f wmma_bf16(v16bf a, v16bf b, v8f c) {
    return __builtin_amdgcn_wmma_f32_16x16x32_bf16(
        false, a, false, b, (short)0, c, false, false);
}

// ---------------------------------------------------------------------------
// GEMM: C[M,N] = act(A[M,K] @ Wbf[N,K]^T + bias) + residual
// A is f32; Wbf is bf16, zero-padded to [Np(mult 128) x Kp(mult 32)].
// 256 threads = 8 waves; block tile 64(M) x 128(N); wave tile 16 x 64.
// B tile filled by async-to-LDS b128 copies (raw bf16, no conversion).
// act: 0=none, 1=gelu(exact), 2=relu.
// ---------------------------------------------------------------------------
__global__ __launch_bounds__(256, 2)
void gemm_wmma_kernel(const float* __restrict__ A, const unsigned short* __restrict__ Wbf,
                      const float* __restrict__ bias, const float* __restrict__ res,
                      float* __restrict__ Cout, int M, int N, int K, int Kp, int act) {
    __shared__ __align__(16) unsigned short As[64][32];
    __shared__ __align__(16) unsigned short Bs[128][32];

    const int bm = blockIdx.x * 64;
    const int bn = blockIdx.y * 128;
    const int t    = threadIdx.x;
    const int wave = t >> 5;
    const int lane = t & 31;
    const int h    = lane >> 4;
    const int l16  = lane & 15;
    const int wm   = (wave >> 1) * 16;  // 4 wave-rows of 16
    const int wn   = (wave & 1) * 64;   // 2 wave-cols of 64 (4 n-tiles)

    v8f acc[4] = {};

    for (int kc = 0; kc < Kp; kc += 32) {
        // ---- async copy B tile: 128 rows x 32 bf16 = 512 x 16B ----
        for (int i = t; i < 512; i += 256) {
            int r = i >> 2, q = i & 3;
            unsigned lds_off = (unsigned)(uintptr_t)&Bs[r][q * 8];
            unsigned long long ga =
                (unsigned long long)(uintptr_t)(Wbf + (size_t)(bn + r) * Kp + kc + q * 8);
            asm volatile("global_load_async_to_lds_b128 %0, %1, off"
                         :: "v"(lds_off), "v"(ga) : "memory");
        }
        // ---- A tile: 64 rows x 32 f32 -> bf16 (vectorized fast path) ----
        for (int i = t; i < 64 * 8; i += 256) {
            int r = i >> 3, c4 = (i & 7) * 4;
            int gm = bm + r, gk = kc + c4;
            uint2 pk;
            if (gm < M && gk + 3 < K) {
                const float4 v = *(const float4*)(A + (size_t)gm * K + gk);
                pk.x = pack_bf2(v.x, v.y);
                pk.y = pack_bf2(v.z, v.w);
            } else {
                float e[4];
                for (int q = 0; q < 4; ++q) {
                    int kk = gk + q;
                    e[q] = (gm < M && kk < K) ? A[(size_t)gm * K + kk] : 0.f;
                }
                pk.x = pack_bf2(e[0], e[1]);
                pk.y = pack_bf2(e[2], e[3]);
            }
            *(uint2*)&As[r][c4] = pk;
        }
        asm volatile("s_wait_asynccnt 0x0" ::: "memory");
        __syncthreads();

        v16bf afrag = load_frag(&As[wm + l16][0], h);
        for (int nt = 0; nt < 4; ++nt) {
            v16bf bfrag = load_frag(&Bs[wn + nt * 16 + l16][0], h);
            acc[nt] = wmma_bf16(afrag, bfrag, acc[nt]);
        }
        __syncthreads();
    }

    for (int nt = 0; nt < 4; ++nt) {
        int gc = bn + wn + nt * 16 + l16;
        if (gc >= N) continue;
        float bv = bias ? bias[gc] : 0.f;
        for (int j = 0; j < 8; ++j) {
            int gm = bm + wm + h * 8 + j;
            if (gm >= M) continue;
            float v = acc[nt][j] + bv;
            if (act == 1)      v = 0.5f * v * (1.f + erff(v * 0.70710678118654752f));
            else if (act == 2) v = fmaxf(v, 0.f);
            if (res) v += res[(size_t)gm * N + gc];
            Cout[(size_t)gm * N + gc] = v;
        }
    }
}

// ---------------------------------------------------------------------------
// Weight pre-conversion: f32 [N,K] -> bf16 padded [Np x Kp], zero filled.
// ---------------------------------------------------------------------------
__global__ void convert_w_kernel(const float* __restrict__ w, unsigned short* __restrict__ dst,
                                 int N, int K, int Kp, int total) {
    int idx = blockIdx.x * 256 + threadIdx.x;
    if (idx >= total) return;
    int r = idx / Kp, c = idx % Kp;
    float v = (r < N && c < K) ? w[(size_t)r * K + c] : 0.f;
    dst[idx] = f2bf(v);
}

// ---------------------------------------------------------------------------
// Fused window attention: one workgroup (4 waves) per (window, head).
// Q,K,V are 49x32 (head_dim==32 everywhere), padded to 64.
// ---------------------------------------------------------------------------
__global__ __launch_bounds__(128, 2)
void attn_wmma_kernel(const float* __restrict__ qkv, const float* __restrict__ rpb,
                      float* __restrict__ out, int C, int nH, int res, int shift,
                      int nWn) {
    __shared__ __align__(16) unsigned short Qs[64][32];
    __shared__ __align__(16) unsigned short Ks[64][32];
    __shared__ __align__(16) unsigned short Vt[32][64];  // [head_dim][key]
    __shared__ __align__(16) unsigned short Ps[64][64];

    const int wi   = blockIdx.x / nH;
    const int head = blockIdx.x % nH;
    const int t    = threadIdx.x;
    const int wave = t >> 5;
    const int lane = t & 31;
    const int h    = lane >> 4;
    const int l16  = lane & 15;
    const int m0   = wave * 16;

    for (int i = t; i < 64 * 32; i += 128) { Qs[i >> 5][i & 31] = 0; Ks[i >> 5][i & 31] = 0; }
    for (int i = t; i < 32 * 64; i += 128) { Vt[i >> 6][i & 63] = 0; }
    __syncthreads();

    const size_t base = (size_t)wi * 49;
    const int C3 = 3 * C;
    const float scale = 0.17677669529663687f;  // 32^-0.5
    for (int i = t; i < 49 * 32; i += 128) {
        int r = i >> 5, d = i & 31;
        const float* rp = qkv + (base + r) * C3 + head * 32 + d;
        Qs[r][d] = f2bf(rp[0] * scale);
        Ks[r][d] = f2bf(rp[C]);
        Vt[d][r] = f2bf(rp[2 * C]);
    }
    __syncthreads();

    // S = Q @ K^T : 4 n-tiles per wave, single K-step (hd=32)
    v16bf aq = load_frag(&Qs[m0 + l16][0], h);
    v8f s[4] = {};
    for (int nt = 0; nt < 4; ++nt) {
        v16bf bk = load_frag(&Ks[nt * 16 + l16][0], h);
        s[nt] = wmma_bf16(aq, bk, s[nt]);
    }

    // rel-pos bias + shift mask
    const int n  = res / 7;
    const int wl = wi % nWn;
    const int wr = wl / n, wc = wl % n;
    for (int nt = 0; nt < 4; ++nt) {
        int jj = nt * 16 + l16;
        bool kv = jj < 49;
        int rj = jj / 7, cj = jj % 7;
        int ck = 0;
        if (shift && kv) {
            int y = wr * 7 + rj, x = wc * 7 + cj;
            ck = ((y >= res - 7) + (y >= res - 3)) * 3 + (x >= res - 7) + (x >= res - 3);
        }
        for (int j = 0; j < 8; ++j) {
            int i = m0 + h * 8 + j;
            float v = s[nt][j];
            if (kv && i < 49) {
                int ri = i / 7, ci = i % 7;
                int rel = (ri - rj + 6) * 13 + (ci - cj + 6);
                v += rpb[rel * nH + head];
                if (shift) {
                    int y = wr * 7 + ri, x = wc * 7 + ci;
                    int cq = ((y >= res - 7) + (y >= res - 3)) * 3 + (x >= res - 7) + (x >= res - 3);
                    if (cq != ck) v -= 100.f;
                }
            } else {
                v = -1e30f;
            }
            s[nt][j] = v;
        }
    }

    // softmax per row (reduce over 4 tiles x 16 lanes)
    for (int j = 0; j < 8; ++j) {
        float m = fmaxf(fmaxf(s[0][j], s[1][j]), fmaxf(s[2][j], s[3][j]));
        for (int o = 1; o < 16; o <<= 1) m = fmaxf(m, __shfl_xor(m, o, 32));
        float sum = 0.f;
        for (int nt = 0; nt < 4; ++nt) {
            float e = __expf(s[nt][j] - m);
            s[nt][j] = e;
            sum += e;
        }
        for (int o = 1; o < 16; o <<= 1) sum += __shfl_xor(sum, o, 32);
        float inv = 1.f / sum;
        for (int nt = 0; nt < 4; ++nt) s[nt][j] *= inv;
    }

    for (int nt = 0; nt < 4; ++nt)
        for (int j = 0; j < 8; ++j)
            Ps[m0 + h * 8 + j][nt * 16 + l16] = f2bf(s[nt][j]);
    __syncthreads();

    // O = P @ V : K=64 (2 chunks), N=32 (2 tiles)
    v8f o[2] = {};
    for (int kc = 0; kc < 2; ++kc) {
        v16bf pa = load_frag(&Ps[m0 + l16][kc * 32], h);
        for (int nt = 0; nt < 2; ++nt) {
            v16bf vb = load_frag(&Vt[nt * 16 + l16][kc * 32], h);
            o[nt] = wmma_bf16(pa, vb, o[nt]);
        }
    }
    for (int nt = 0; nt < 2; ++nt) {
        int col = head * 32 + nt * 16 + l16;
        for (int j = 0; j < 8; ++j) {
            int i = m0 + h * 8 + j;
            if (i < 49) out[(base + i) * C + col] = o[nt][j];
        }
    }
}

// ---------------------------------------------------------------------------
// LayerNorm over last dim C (<=4096), one 256-thread block per row (in==out ok)
// ---------------------------------------------------------------------------
__global__ __launch_bounds__(256)
void ln_kernel(const float* __restrict__ in, float* __restrict__ out,
               const float* __restrict__ g, const float* __restrict__ b, int C) {
    __shared__ float red[256];
    const size_t row = blockIdx.x;
    const float* xr = in + row * C;
    float v[16];
    int cnt = 0;
    float s = 0.f;
    for (int c = threadIdx.x; c < C; c += 256) { float t = xr[c]; v[cnt++] = t; s += t; }
    red[threadIdx.x] = s; __syncthreads();
    for (int o = 128; o > 0; o >>= 1) { if (threadIdx.x < o) red[threadIdx.x] += red[threadIdx.x + o]; __syncthreads(); }
    float mu = red[0] / C; __syncthreads();
    s = 0.f;
    for (int i = 0; i < cnt; ++i) { float d = v[i] - mu; s += d * d; }
    red[threadIdx.x] = s; __syncthreads();
    for (int o = 128; o > 0; o >>= 1) { if (threadIdx.x < o) red[threadIdx.x] += red[threadIdx.x + o]; __syncthreads(); }
    float rs = rsqrtf(red[0] / C + 1e-5f);
    cnt = 0;
    for (int c = threadIdx.x; c < C; c += 256)
        out[row * C + c] = (v[cnt++] - mu) * rs * g[c] + b[c];
}

// ---------------------------------------------------------------------------
// Elementwise / gather kernels
// ---------------------------------------------------------------------------
__global__ void patchify_kernel(const float* __restrict__ x, float* __restrict__ out, int total) {
    int idx = blockIdx.x * 256 + threadIdx.x;
    if (idx >= total) return;
    int f = idx % 48, tok = idx / 48;
    int b = tok / 3136, p = tok % 3136;
    int i = p / 56, j = p % 56;
    int c = f >> 4, dy = (f >> 2) & 3, dx = f & 3;
    out[idx] = x[(((size_t)(b * 3 + c) * 224) + (i * 4 + dy)) * 224 + (j * 4 + dx)];
}

__global__ void win_part_kernel(const float* __restrict__ X, float* __restrict__ out,
                                int res, int C, int shift, int total) {
    int idx = blockIdx.x * 256 + threadIdx.x;
    if (idx >= total) return;
    int c = idx % C, r = idx / C;
    int p = r % 49, w = r / 49;
    int n = res / 7, nWn = n * n;
    int b = w / nWn, wl = w % nWn;
    int wr = wl / n, wc = wl % n;
    int a = p / 7, bb = p % 7;
    int y = wr * 7 + a + shift;  if (y >= res) y -= res;
    int x = wc * 7 + bb + shift; if (x >= res) x -= res;
    out[idx] = X[((size_t)b * res * res + y * res + x) * C + c];
}

__global__ void win_rev_add_kernel(float* __restrict__ X, const float* __restrict__ Pw,
                                   int res, int C, int shift, int total) {
    int idx = blockIdx.x * 256 + threadIdx.x;
    if (idx >= total) return;
    int c = idx % C, r = idx / C;
    int b = r / (res * res), pix = r % (res * res);
    int y = pix / res, x = pix % res;
    int yr = y - shift; if (yr < 0) yr += res;
    int xr = x - shift; if (xr < 0) xr += res;
    int n = res / 7;
    int wr = yr / 7, wc = xr / 7;
    int p = (yr % 7) * 7 + (xr % 7);
    X[idx] += Pw[(((size_t)(b * n * n + wr * n + wc)) * 49 + p) * C + c];
}

__global__ void merge_gather_kernel(const float* __restrict__ X, float* __restrict__ out,
                                    int res, int C, int total) {
    int idx = blockIdx.x * 256 + threadIdx.x;
    if (idx >= total) return;
    int C4 = 4 * C;
    int f = idx % C4, r = idx / C4;
    int hres = res / 2;
    int b = r / (hres * hres), p = r % (hres * hres);
    int oi = p / hres, oj = p % hres;
    int q = f / C, c = f % C;
    int y = oi * 2 + (q & 1), x = oj * 2 + (q >> 1);
    out[idx] = X[((size_t)b * res * res + y * res + x) * C + c];
}

__global__ void pool_kernel(const float* __restrict__ in, float* __restrict__ out, int L, int C) {
    int idx = blockIdx.x * 256 + threadIdx.x;
    if (idx >= 32 * C) return;
    int b = idx / C, c = idx % C;
    float s = 0.f;
    for (int i = 0; i < L; ++i) s += in[((size_t)b * L + i) * C + c];
    out[idx] = s / (float)L;
}

__global__ void head_softmax_kernel(const float* __restrict__ logits, float* __restrict__ out) {
    int b = threadIdx.x;
    if (b >= 32) return;
    float a = logits[b * 2], c = logits[b * 2 + 1];
    float m = fmaxf(a, c);
    float ea = __expf(a - m), ec = __expf(c - m);
    float inv = 1.f / (ea + ec);
    out[b * 2] = ea * inv;  out[b * 2 + 1] = ec * inv;
    out[64 + b * 2] = ea * inv;  out[64 + b * 2 + 1] = ec * inv;
}

// ---------------------------------------------------------------------------
// Host orchestration
// ---------------------------------------------------------------------------
struct WB { const unsigned short* w; int Kp; };

static inline unsigned ceil_div(long long a, long long b) { return (unsigned)((a + b - 1) / b); }

static void gemm(hipStream_t st, const float* A, WB w, const float* bias,
                 const float* res, float* C, int M, int N, int K, int act) {
    dim3 g(ceil_div(M, 64), ceil_div(N, 128));
    gemm_wmma_kernel<<<g, 256, 0, st>>>(A, w.w, bias, res, C, M, N, K, w.Kp, act);
}

extern "C" void kernel_launch(void* const* d_in, const int* in_sizes, int n_in,
                              void* d_out, int out_size, void* d_ws, size_t ws_size,
                              hipStream_t stream) {
    static const int DIMS[4]   = {128, 256, 512, 1024};
    static const int DEPTHS[4] = {2, 2, 18, 2};
    static const int HEADS[4]  = {4, 8, 16, 32};
    static const int RESA[4]   = {56, 28, 14, 7};
    const int B = 32;

    // ---- flatten inputs in setup_inputs() dict order ----
    const float* x_img = (const float*)d_in[0];
    int pi = 1;
    auto P = [&]() -> const float* { return (const float*)d_in[pi++]; };

    const float* pe_w = P();  const float* pe_b = P();
    const float* pe_ln_g = P(); const float* pe_ln_b = P();
    const float* norm_g = P();  const float* norm_b = P();
    const float* h1_w = P();  const float* h1_b = P();
    const float* h2_w = P();  const float* h2_b = P();

    struct Blk { const float *ln1_g,*ln1_b,*qkv_w,*qkv_b,*rpb,*proj_w,*proj_b,
                             *ln2_g,*ln2_b,*fc1_w,*fc1_b,*fc2_w,*fc2_b; };
    struct Mrg { const float *ln_g,*ln_b,*red_w; };
    Blk blk[4][18];
    Mrg mrg[3];
    for (int s = 0; s < 4; ++s) {
        for (int j = 0; j < DEPTHS[s]; ++j) {
            Blk& bb = blk[s][j];
            bb.ln1_g = P(); bb.ln1_b = P();
            bb.qkv_w = P(); bb.qkv_b = P();
            bb.rpb   = P();
            bb.proj_w = P(); bb.proj_b = P();
            bb.ln2_g = P(); bb.ln2_b = P();
            bb.fc1_w = P(); bb.fc1_b = P();
            bb.fc2_w = P(); bb.fc2_b = P();
        }
        if (s < 3) { mrg[s].ln_g = P(); mrg[s].ln_b = P(); mrg[s].red_w = P(); }
    }

    // ---- workspace layout ----
    char* ws = (char*)d_ws;
    const size_t MB = 1ull << 20;
    float* X  = (float*)(ws + 0);         //  52 MB activations
    float* Y  = (float*)(ws + 52 * MB);   //  52 MB activations (double buffer)
    float* T1 = (float*)(ws + 104 * MB);  //  52 MB LN out / attn out / merge gather
    float* T2 = (float*)(ws + 156 * MB);  // 208 MB qkv / fc1 out / merge LN out
    float* T3 = (float*)(ws + 364 * MB);  //  52 MB windows / proj out / patchify
    float* S0 = (float*)(ws + 416 * MB);
    float* S1 = (float*)(ws + 417 * MB);
    float* S2 = (float*)(ws + 418 * MB);
    size_t wcur = 420 * MB;               // bf16 weight arena (~180 MB)

    auto convw = [&](const float* w, int N, int K) -> WB {
        int Np = (N + 127) & ~127;
        int Kp = (K + 31) & ~31;
        unsigned short* d = (unsigned short*)(ws + wcur);
        size_t bytes = (size_t)Np * Kp * 2;
        wcur += (bytes + 255) & ~(size_t)255;
        int total = Np * Kp;
        convert_w_kernel<<<ceil_div(total, 256), 256, 0, stream>>>(w, d, N, K, Kp, total);
        return WB{d, Kp};
    };

    // ---- pre-convert all weights to padded bf16 (once per launch) ----
    WB peW = convw(pe_w, 128, 48);
    WB h1W = convw(h1_w, 512, 1024);
    WB h2W = convw(h2_w, 2, 512);
    WB qkvW[4][18], projW[4][18], fc1W[4][18], fc2W[4][18], redW[3];
    for (int s = 0; s < 4; ++s) {
        int C = DIMS[s];
        for (int j = 0; j < DEPTHS[s]; ++j) {
            qkvW[s][j] = convw(blk[s][j].qkv_w, 3 * C, C);
            projW[s][j] = convw(blk[s][j].proj_w, C, C);
            fc1W[s][j] = convw(blk[s][j].fc1_w, 4 * C, C);
            fc2W[s][j] = convw(blk[s][j].fc2_w, C, 4 * C);
        }
        if (s < 3) redW[s] = convw(mrg[s].red_w, 2 * C, 4 * C);
    }

    // ---- patch embed ----
    {
        int Btok = B * 3136;
        int total = Btok * 48;
        patchify_kernel<<<ceil_div(total, 256), 256, 0, stream>>>(x_img, T3, total);
        gemm(stream, T3, peW, pe_b, nullptr, X, Btok, 128, 48, 0);
        ln_kernel<<<Btok, 256, 0, stream>>>(X, X, pe_ln_g, pe_ln_b, 128);
    }

    // ---- stages ----
    for (int s = 0; s < 4; ++s) {
        const int res = RESA[s], C = DIMS[s], nH = HEADS[s];
        const int L = res * res, Bt = B * L;
        const int n = res / 7, nWn = n * n;
        for (int j = 0; j < DEPTHS[s]; ++j) {
            const Blk& p = blk[s][j];
            const int shift = (j % 2 == 1 && res > 7) ? 3 : 0;
            const int totBC = Bt * C;

            ln_kernel<<<Bt, 256, 0, stream>>>(X, T1, p.ln1_g, p.ln1_b, C);
            win_part_kernel<<<ceil_div(totBC, 256), 256, 0, stream>>>(T1, T3, res, C, shift, totBC);
            gemm(stream, T3, qkvW[s][j], p.qkv_b, nullptr, T2, Bt, 3 * C, C, 0);
            attn_wmma_kernel<<<dim3((unsigned)(B * nWn * nH)), 128, 0, stream>>>(
                T2, p.rpb, T1, C, nH, res, shift, nWn);
            gemm(stream, T1, projW[s][j], p.proj_b, nullptr, T3, Bt, C, C, 0);
            win_rev_add_kernel<<<ceil_div(totBC, 256), 256, 0, stream>>>(X, T3, res, C, shift, totBC);
            ln_kernel<<<Bt, 256, 0, stream>>>(X, T1, p.ln2_g, p.ln2_b, C);
            gemm(stream, T1, fc1W[s][j], p.fc1_b, nullptr, T2, Bt, 4 * C, C, 1 /*gelu*/);
            gemm(stream, T2, fc2W[s][j], p.fc2_b, X /*residual*/, Y, Bt, C, 4 * C, 0);
            { float* tmp = X; X = Y; Y = tmp; }
        }
        if (s < 3) {
            const Mrg& m = mrg[s];
            const int rows = B * (L / 4);
            const int total = rows * 4 * C;
            merge_gather_kernel<<<ceil_div(total, 256), 256, 0, stream>>>(X, T1, res, C, total);
            ln_kernel<<<rows, 256, 0, stream>>>(T1, T2, m.ln_g, m.ln_b, 4 * C);
            gemm(stream, T2, redW[s], nullptr, nullptr, Y, rows, 2 * C, 4 * C, 0);
            { float* tmp = X; X = Y; Y = tmp; }
        }
    }

    // ---- head ----
    ln_kernel<<<B * 49, 256, 0, stream>>>(X, T1, norm_g, norm_b, 1024);
    pool_kernel<<<ceil_div(B * 1024, 256), 256, 0, stream>>>(T1, S0, 49, 1024);
    gemm(stream, S0, h1W, h1_b, nullptr, S1, B, 512, 1024, 2 /*relu*/);
    gemm(stream, S1, h2W, h2_b, nullptr, S2, B, 2, 512, 0);
    head_softmax_kernel<<<1, 32, 0, stream>>>(S2, (float*)d_out);
}
